// GCN_54941221650950
// MI455X (gfx1250) — matrix operand order repaired and checked by reading, not compiled, not run on hardware
//
#include <hip/hip_runtime.h>
#include <hip/hip_bf16.h>

typedef __attribute__((ext_vector_type(16))) __bf16 v16bf;
typedef __attribute__((ext_vector_type(8)))  float  v8f;

#define NN   50000   // nodes
#define EE   800000  // edges
#define DM   128     // feature dim (both layers)
#define GG   64      // graphs
#define CC   10      // classes

// ---------------------------------------------------------------------------
// init: deg = 1 (self loop), pool = 0, cnt = 0
// ---------------------------------------------------------------------------
__global__ void k_init(float* __restrict__ deg, float* __restrict__ pool,
                       float* __restrict__ cnt) {
    int t = blockIdx.x * blockDim.x + threadIdx.x;
    if (t < NN) deg[t] = 1.0f;
    if (t < GG * DM) pool[t] = 0.0f;
    if (t < GG) cnt[t] = 0.0f;
}

// deg[col[e]] += 1
__global__ void k_degree(const int* __restrict__ ei, float* __restrict__ deg) {
    int e = blockIdx.x * blockDim.x + threadIdx.x;
    if (e < EE) atomicAdd(&deg[ei[EE + e]], 1.0f);
}

__global__ void k_rsqrt(const float* __restrict__ deg, float* __restrict__ dis) {
    int t = blockIdx.x * blockDim.x + threadIdx.x;
    if (t < NN) dis[t] = rsqrtf(deg[t]);
}

// ---------------------------------------------------------------------------
// hs = (X @ W) * dis[:,None]; acc = hs  (self-loop initialization)
// One block per 16-row strip; wave w owns 16-col tile w. bf16 WMMA, f32 acc.
// W is staged once per block into LDS, bf16, pre-swizzled into B-fragment
// order so each fragment read is one aligned 32B ds load.
// ---------------------------------------------------------------------------
__global__ __launch_bounds__(256)
void k_gemm_scale(const float* __restrict__ X, const float* __restrict__ W,
                  const float* __restrict__ dis, float* __restrict__ hs,
                  float* __restrict__ acc) {
    // sB[kt][nt][lane][i] : element i of lane `lane` for column tile nt, K tile kt
    __shared__ __align__(32) __bf16 sB[4 * 8 * 32 * 16];   // 32 KB

    const int tid = threadIdx.x;
    // Stage W (128x128 f32, row-major [K][N]) -> LDS bf16 fragments.
    // Linear (coalesced) global reads; swizzled LDS writes.
    #pragma unroll
    for (int u = 0; u < 64; ++u) {
        const int wi  = tid * 64 + u;        // 0 .. 16383, contiguous per thread
        const int k   = wi >> 7;             // row of W (K index)
        const int col = wi & (DM - 1);       // column of W (N index)
        const int kt  = k >> 5;
        const int kr  = k & 31;
        const int nt  = col >> 4;
        const int lane = (col & 15) + ((kr >= 16) ? 16 : 0);
        const int i    = kr & 15;
        sB[((kt * 8 + nt) * 32 + lane) * 16 + i] = (__bf16)W[wi];
    }
    __syncthreads();

    const int wave = tid >> 5;                // 0..7 -> column tile
    const int lane = tid & 31;
    const int m0   = blockIdx.x * 16;
    const int mr   = lane & 15;
    const bool hi  = lane >= 16;
    const int row  = m0 + mr;
    const int col  = wave * 16 + mr;          // output column for this lane

    v8f c = {};
    #pragma unroll
    for (int kt = 0; kt < 4; ++kt) {
        // A fragment: 16x32 bf16 in contiguous k order (k = 16h + s), same as sB.
        // lanes 0-15: elems 0..15 = K kt*32+0..15
        // lanes 16-31: elems 0..15 = K kt*32+16..31
        const int kbA = kt * 32 + (hi ? 16 : 0);
        const float* ap = X + row * DM + kbA;
        v16bf a;
        #pragma unroll
        for (int i = 0; i < 8; ++i) a[i] = (__bf16)ap[i];
        #pragma unroll
        for (int i = 0; i < 8; ++i) a[8 + i] = (__bf16)ap[8 + i];

        // B fragment: one aligned 32-byte LDS read (2 x ds_load_b128)
        const v16bf b = *(const v16bf*)&sB[((kt * 8 + wave) * 32 + lane) * 16];

        c = __builtin_amdgcn_wmma_f32_16x16x32_bf16(
                false, a, false, b, (short)0, c, false, false);
    }

    // C/D layout: VGPR r holds row m0 + r (+8 for lanes 16-31), column `col`
    #pragma unroll
    for (int r = 0; r < 8; ++r) {
        const int gr  = m0 + r + (hi ? 8 : 0);
        const float v = c[r] * dis[gr];
        const int idx = gr * DM + col;
        hs[idx]  = v;
        acc[idx] = v;
    }
}

// ---------------------------------------------------------------------------
// Per-edge scatter: acc[col] += hs[row].  One wave32 per edge, 4 floats/lane.
// Accumulator buffers are L2-resident (192 MB >> 77 MB working set).
// ---------------------------------------------------------------------------
__global__ __launch_bounds__(256)
void k_scatter(const int* __restrict__ ei, const float* __restrict__ hs,
               float* __restrict__ acc) {
    const int t    = blockIdx.x * blockDim.x + threadIdx.x;
    const int e    = t >> 5;
    const int lane = t & 31;
    if (e >= EE) return;
    const int r = ei[e];          // source
    const int c = ei[EE + e];     // target
    const float4 m = *(const float4*)(hs + r * DM + lane * 4);
    float* dst = acc + c * DM + lane * 4;
    atomicAdd(dst + 0, m.x);
    atomicAdd(dst + 1, m.y);
    atomicAdd(dst + 2, m.z);
    atomicAdd(dst + 3, m.w);
}

// out[v,j] = relu(dis[v]*acc[v,j] + b[j])
__global__ void k_bias_relu(const float* __restrict__ acc,
                            const float* __restrict__ dis,
                            const float* __restrict__ b,
                            float* __restrict__ out) {
    int t = blockIdx.x * blockDim.x + threadIdx.x;
    if (t >= NN * DM) return;
    int v = t >> 7, j = t & (DM - 1);
    float x = dis[v] * acc[t] + b[j];
    out[t] = x > 0.0f ? x : 0.0f;
}

// pool[batch[v]] += h[v]; cnt[batch[v]] += 1
__global__ void k_pool(const int* __restrict__ batch, const float* __restrict__ h,
                       float* __restrict__ pool, float* __restrict__ cnt) {
    int t = blockIdx.x * blockDim.x + threadIdx.x;
    if (t >= NN * DM) return;
    int v = t >> 7, j = t & (DM - 1);
    int g = batch[v];
    atomicAdd(&pool[g * DM + j], h[t]);
    if (j == 0) atomicAdd(&cnt[g], 1.0f);
}

// out[g,c] = bc[c] + (pool[g]/max(cnt[g],1)) . Wc[:,c]
__global__ void k_classify(const float* __restrict__ pool, const float* __restrict__ cnt,
                           const float* __restrict__ Wc, const float* __restrict__ bc,
                           float* __restrict__ out) {
    int t = blockIdx.x * blockDim.x + threadIdx.x;
    if (t >= GG * CC) return;
    int g = t / CC, c = t % CC;
    float inv = 1.0f / fmaxf(cnt[g], 1.0f);
    float s = bc[c];
    for (int j = 0; j < DM; ++j)
        s += pool[g * DM + j] * inv * Wc[j * CC + c];
    out[t] = s;
}

extern "C" void kernel_launch(void* const* d_in, const int* in_sizes, int n_in,
                              void* d_out, int out_size, void* d_ws, size_t ws_size,
                              hipStream_t stream) {
    const float* x   = (const float*)d_in[0];
    const int*   ei  = (const int*)d_in[1];
    const int*   bat = (const int*)d_in[2];
    const float* W1  = (const float*)d_in[3];
    const float* b1  = (const float*)d_in[4];
    const float* W2  = (const float*)d_in[5];
    const float* b2  = (const float*)d_in[6];
    const float* Wc  = (const float*)d_in[7];
    const float* bc  = (const float*)d_in[8];
    float* out = (float*)d_out;

    const size_t NH = (size_t)NN * DM;
    float* hs   = (float*)d_ws;
    float* acc  = hs + NH;
    float* buf1 = acc + NH;
    float* deg  = buf1 + NH;
    float* dis  = deg + NN;
    float* pool = dis + NN;
    float* cnt  = pool + (size_t)GG * DM;

    const int T = 256;
    // degree + normalization
    k_init  <<<(NN + T - 1) / T, T, 0, stream>>>(deg, pool, cnt);
    k_degree<<<(EE + T - 1) / T, T, 0, stream>>>(ei, deg);
    k_rsqrt <<<(NN + T - 1) / T, T, 0, stream>>>(deg, dis);

    const int gemmBlocks = NN / 16;                 // 3125
    const int scatBlocks = (EE * 32 + T - 1) / T;   // 100000
    const int ewBlocks   = (NN * DM + T - 1) / T;   // 25000

    // Layer 1
    k_gemm_scale<<<gemmBlocks, T, 0, stream>>>(x, W1, dis, hs, acc);
    k_scatter   <<<scatBlocks, T, 0, stream>>>(ei, hs, acc);
    k_bias_relu <<<ewBlocks,   T, 0, stream>>>(acc, dis, b1, buf1);

    // Layer 2
    k_gemm_scale<<<gemmBlocks, T, 0, stream>>>(buf1, W2, dis, hs, acc);
    k_scatter   <<<scatBlocks, T, 0, stream>>>(ei, hs, acc);
    k_bias_relu <<<ewBlocks,   T, 0, stream>>>(acc, dis, b2, buf1);

    // Pool + classify
    k_pool    <<<ewBlocks, T, 0, stream>>>(bat, buf1, pool, cnt);
    k_classify<<<(GG * CC + T - 1) / T, T, 0, stream>>>(pool, cnt, Wc, bc, out);
}
